// TemporalMoEBlock_23553600651635
// MI455X (gfx1250) — compile-verified, matrix operand order checked by reading, not run on hardware
//
#include <hip/hip_runtime.h>
#include <hip/hip_bf16.h>
#include <math.h>

typedef __attribute__((ext_vector_type(16))) _Float16 v16h;
typedef __attribute__((ext_vector_type(8)))  _Float16 v8h;
typedef __attribute__((ext_vector_type(8)))  float    v8f;
typedef __attribute__((ext_vector_type(4)))  int      v4i;

#define DEV static __device__ __forceinline__

#define TKN  4096   // tokens = B*S
#define DM   768
#define NH   12
#define HD   64
#define FF   3072
#define NE   8
#define NQKV 2304
#define SEQ  1024
#define NB   4

// ---------------- async-copy path (gfx1250 GLOBAL_LOAD_ASYNC_TO_LDS) --------

#if __has_builtin(__builtin_amdgcn_global_load_async_to_lds_b128) && \
    __has_builtin(__builtin_amdgcn_s_wait_asynccnt)
#define HAVE_ASYNC 1
#else
#define HAVE_ASYNC 0
#endif

#if HAVE_ASYNC
#define ASYNC_WAIT(n) __builtin_amdgcn_s_wait_asynccnt(n)
#else
#define ASYNC_WAIT(n)
#endif

DEV void async_copy16(const _Float16* g, _Float16* l) {
#if HAVE_ASYNC
  // ROCm clang-22 signature takes generic v4i* operands:
  // (global source VADDR, LDS destination VDST, imm offset, cpol)
  __builtin_amdgcn_global_load_async_to_lds_b128((v4i*)g, (v4i*)l, 0, 0);
#else
  *(v8h*)l = *(const v8h*)g;   // sync fallback: global_load_b128 + ds_store_b128
#endif
}

// ---------------- WMMA helpers (gfx1250 wave32, 16x16x32 f16 -> f32) --------

DEV v16h load_frag16(const _Float16* p) {
  // per-lane A/B fragment: elements 0..7 = K k0..k0+7, elements 8..15 = K k0+16..k0+23
  v8h lo = *(const v8h*)(p);
  v8h hi = *(const v8h*)(p + 16);
  return __builtin_shufflevector(lo, hi, 0,1,2,3,4,5,6,7,8,9,10,11,12,13,14,15);
}

DEV v8f wmma16(v16h a, v16h b, v8f c) {
  return __builtin_amdgcn_wmma_f32_16x16x32_f16(false, a, false, b, (short)0, c,
                                                false, false);
}

DEV float gelu_tanh(float x) {
  float x3 = x * x * x;
  return 0.5f * x * (1.0f + tanhf(0.7978845608028654f * (x + 0.044715f * x3)));
}

// ---------------- LDS-staged, double-buffered GEMM core ----------------------
// Block tile: M=128 x N=256, K-step 32. 8 waves, each computing a 64x64 tile.
// LDS rows padded to 40 halves (80 B): ds_load_b128 fragment reads are
// bank-conflict-free (row stride 20 DW, gcd(20,64)=4 -> 16 distinct banks).

#define LDA   40
#define ASLAB (128 * LDA)   // halves
#define BSLAB (256 * LDA)   // halves
#define SMEM_HALVES (2 * ASLAB + 2 * BSLAB)

// Issue one stage: A slab 128x32 (512 x 16B chunks), B slab 256x32 (1024 chunks),
// spread over 256 threads -> 2 + 4 async b128 copies per thread.
DEV void stage_issue(const _Float16* A, const _Float16* BT, int K,
                     int m0, int n0, int kk,
                     _Float16* As, _Float16* Bs, int tid) {
  #pragma unroll
  for (int t = 0; t < 2; t++) {
    const int c = t * 256 + tid;
    const int row = c >> 2, q = (c & 3) * 8;
    async_copy16(A + (size_t)(m0 + row) * K + kk + q, As + row * LDA + q);
  }
  #pragma unroll
  for (int t = 0; t < 4; t++) {
    const int c = t * 256 + tid;
    const int row = c >> 2, q = (c & 3) * 8;
    async_copy16(BT + (size_t)(n0 + row) * K + kk + q, Bs + row * LDA + q);
  }
}

DEV void gemm_core(const _Float16* A, const _Float16* BT, int K,
                   int m0, int n0, _Float16* smem, v8f acc[4][4]) {
  const int tid  = threadIdx.x;
  const int lane = tid & 31;
  const int wave = tid >> 5;
  const int rl = lane & 15, ko = (lane >> 4) << 3;
  const int mw = (wave & 1) * 64, nw = (wave >> 1) * 64;
  _Float16* As0 = smem;
  _Float16* As1 = smem + ASLAB;
  _Float16* Bs0 = smem + 2 * ASLAB;
  _Float16* Bs1 = smem + 2 * ASLAB + BSLAB;

  #pragma unroll
  for (int i = 0; i < 4; i++)
    #pragma unroll
    for (int j = 0; j < 4; j++)
      #pragma unroll
      for (int r = 0; r < 8; r++) acc[i][j][r] = 0.0f;

  const int nst = K >> 5;
  stage_issue(A, BT, K, m0, n0, 0, As0, Bs0, tid);
  for (int s = 0; s < nst; s++) {
    _Float16* Ac = (s & 1) ? As1 : As0;
    _Float16* Bc = (s & 1) ? Bs1 : Bs0;
    if (s + 1 < nst) {
      // prefetch next stage while this one computes
      stage_issue(A, BT, K, m0, n0, (s + 1) << 5,
                  (s & 1) ? As0 : As1, (s & 1) ? Bs0 : Bs1, tid);
      ASYNC_WAIT(6);   // 6 newest (next stage) may remain in flight
    } else {
      ASYNC_WAIT(0);
    }
    __syncthreads();   // all waves' current-stage LDS data visible
    v16h a[4], b[4];
    #pragma unroll
    for (int i = 0; i < 4; i++)
      a[i] = load_frag16(Ac + (mw + i * 16 + rl) * LDA + ko);
    #pragma unroll
    for (int j = 0; j < 4; j++)
      b[j] = load_frag16(Bc + (nw + j * 16 + rl) * LDA + ko);
    #pragma unroll
    for (int i = 0; i < 4; i++)
      #pragma unroll
      for (int j = 0; j < 4; j++)
        acc[i][j] = wmma16(a[i], b[j], acc[i][j]);
    __syncthreads();   // all waves done reading before buffer is overwritten
  }
}

// ---------------- conversion kernels ---------------------------------------

__global__ __launch_bounds__(256) void k_f32_to_f16(const float* __restrict__ in,
                                                    _Float16* __restrict__ out,
                                                    int n) {
  int i = blockIdx.x * 256 + threadIdx.x;
  if (i < n) out[i] = (_Float16)in[i];
}

// in: f32 [R,C] row-major -> out: f16 [C,R] row-major (i.e. transposed)
__global__ __launch_bounds__(256) void k_transpose_f16(const float* __restrict__ in,
                                                       _Float16* __restrict__ out,
                                                       int R, int C) {
  int i = blockIdx.x * 256 + threadIdx.x;
  if (i >= R * C) return;
  int c = i / R;
  int r = i - c * R;
  out[i] = (_Float16)in[(size_t)r * C + c];
}

// ---------------- QKV projection + head scatter -----------------------------

__global__ __launch_bounds__(256) void k_gemm_qkv(const _Float16* __restrict__ A,
                                                  const _Float16* __restrict__ BT,
                                                  const float* __restrict__ bias,
                                                  _Float16* __restrict__ qh,
                                                  _Float16* __restrict__ kh,
                                                  _Float16* __restrict__ vT) {
  __shared__ _Float16 smem[SMEM_HALVES];
  const int lane = threadIdx.x & 31, wave = threadIdx.x >> 5;
  const int m0 = blockIdx.y * 128;
  const int n0 = blockIdx.x * 256;
  v8f acc[4][4];
  gemm_core(A, BT, DM, m0, n0, smem, acc);
  const int m0w = m0 + (wave & 1) * 64;
  const int n0w = n0 + (wave >> 1) * 64;
  const int rl = lane & 15, rh = (lane >> 4) << 3;
  #pragma unroll
  for (int j = 0; j < 4; j++) {
    const int c = n0w + j * 16 + rl;
    const float bv = bias[c];
    const int sel = c / DM;
    const int cc = c - sel * DM;
    const int h = cc >> 6, dd = cc & 63;
    #pragma unroll
    for (int i = 0; i < 4; i++)
      #pragma unroll
      for (int r = 0; r < 8; r++) {
        const int t = m0w + i * 16 + r + rh;
        const int b = t >> 10, s = t & 1023;
        const int bh = b * NH + h;
        const float v = acc[i][j][r] + bv;
        if (sel == 0)      qh[((size_t)(bh * SEQ + s)) * HD + dd] = (_Float16)(v * 0.125f);
        else if (sel == 1) kh[((size_t)(bh * SEQ + s)) * HD + dd] = (_Float16)v;
        else               vT[((size_t)(bh * HD + dd)) * SEQ + s] = (_Float16)v;
      }
  }
}

// ---------------- flash attention (one wave = 16 q rows, online softmax) ----

__global__ __launch_bounds__(256) void k_attention(const _Float16* __restrict__ qh,
                                                   const _Float16* __restrict__ kh,
                                                   const _Float16* __restrict__ vT,
                                                   _Float16* __restrict__ ctxh) {
  __shared__ _Float16 lds_p[8][16 * 32];
  const int lane = threadIdx.x & 31, wave = threadIdx.x >> 5;
  const int bh = blockIdx.x;                    // b*12 + h
  const int q0 = blockIdx.y * 128 + wave * 16;  // q tile base
  const int rl = lane & 15, ko = (lane >> 4) << 3, rh = (lane >> 4) << 3;
  _Float16* myp = lds_p[wave];

  const _Float16* qrow = qh + ((size_t)(bh * SEQ) + q0 + rl) * HD + ko;
  const v16h aq0 = load_frag16(qrow);
  const v16h aq1 = load_frag16(qrow + 32);

  v8f accv[4];
  float mrow[8], lrow[8];
  #pragma unroll
  for (int jj = 0; jj < 4; jj++)
    #pragma unroll
    for (int r = 0; r < 8; r++) accv[jj][r] = 0.0f;
  #pragma unroll
  for (int r = 0; r < 8; r++) { mrow[r] = -1e30f; lrow[r] = 0.0f; }

  for (int kv = 0; kv < SEQ; kv += 32) {
    v8f st[2];
    #pragma unroll
    for (int j = 0; j < 2; j++) {
      #pragma unroll
      for (int r = 0; r < 8; r++) st[j][r] = 0.0f;
      const _Float16* krow = kh + ((size_t)(bh * SEQ) + kv + j * 16 + rl) * HD + ko;
      st[j] = wmma16(aq0, load_frag16(krow), st[j]);
      st[j] = wmma16(aq1, load_frag16(krow + 32), st[j]);
    }
    float scl[8];
    #pragma unroll
    for (int r = 0; r < 8; r++) {
      float v = fmaxf(st[0][r], st[1][r]);
      v = fmaxf(v, __shfl_xor(v, 1));
      v = fmaxf(v, __shfl_xor(v, 2));
      v = fmaxf(v, __shfl_xor(v, 4));
      v = fmaxf(v, __shfl_xor(v, 8));
      const float mn = fmaxf(mrow[r], v);
      const float sc = __expf(mrow[r] - mn);
      const float p0 = __expf(st[0][r] - mn);
      const float p1 = __expf(st[1][r] - mn);
      st[0][r] = p0; st[1][r] = p1;
      float ps = p0 + p1;
      ps += __shfl_xor(ps, 1);
      ps += __shfl_xor(ps, 2);
      ps += __shfl_xor(ps, 4);
      ps += __shfl_xor(ps, 8);
      lrow[r] = lrow[r] * sc + ps;
      mrow[r] = mn;
      scl[r] = sc;
    }
    #pragma unroll
    for (int jj = 0; jj < 4; jj++)
      #pragma unroll
      for (int r = 0; r < 8; r++) accv[jj][r] *= scl[r];
    // re-stripe P: C-layout -> A-fragment layout via LDS
    #pragma unroll
    for (int j = 0; j < 2; j++)
      #pragma unroll
      for (int r = 0; r < 8; r++)
        myp[(r + rh) * 32 + j * 16 + rl] = (_Float16)st[j][r];
    asm volatile("s_wait_dscnt 0" ::: "memory");
    const v16h ap = load_frag16(myp + rl * 32 + ko);
    #pragma unroll
    for (int jj = 0; jj < 4; jj++) {
      const _Float16* vrow = vT + ((size_t)(bh * HD) + jj * 16 + rl) * SEQ + kv + ko;
      accv[jj] = wmma16(ap, load_frag16(vrow), accv[jj]);
    }
  }
  const int b = bh / NH, h = bh - b * NH;
  #pragma unroll
  for (int jj = 0; jj < 4; jj++)
    #pragma unroll
    for (int r = 0; r < 8; r++) {
      const int row = q0 + r + rh;
      const int t = (b << 10) + row;
      const int dd = jj * 16 + rl;
      ctxh[(size_t)t * DM + h * HD + dd] = (_Float16)(accv[jj][r] / lrow[r]);
    }
}

// ---------------- generic GEMM -> f16 (optional GELU) -----------------------

__global__ __launch_bounds__(256) void k_gemm_f16out(const _Float16* __restrict__ A,
                                                     const _Float16* __restrict__ BT,
                                                     const float* __restrict__ bias,
                                                     _Float16* __restrict__ C,
                                                     int N, int K, int act) {
  __shared__ _Float16 smem[SMEM_HALVES];
  const int lane = threadIdx.x & 31, wave = threadIdx.x >> 5;
  const int m0 = blockIdx.y * 128;
  const int n0 = blockIdx.x * 256;
  v8f acc[4][4];
  gemm_core(A, BT, K, m0, n0, smem, acc);
  const int m0w = m0 + (wave & 1) * 64;
  const int n0w = n0 + (wave >> 1) * 64;
  const int rl = lane & 15, rh = (lane >> 4) << 3;
  #pragma unroll
  for (int j = 0; j < 4; j++) {
    const int n = n0w + j * 16 + rl;
    const float bv = bias[n];
    #pragma unroll
    for (int i = 0; i < 4; i++)
      #pragma unroll
      for (int r = 0; r < 8; r++) {
        const int m = m0w + i * 16 + r + rh;
        float v = acc[i][j][r] + bv;
        if (act) v = gelu_tanh(v);
        C[(size_t)m * N + n] = (_Float16)v;
      }
  }
}

// ---------------- FFN2 GEMM with gate-weighted accumulate into d_out --------

__global__ __launch_bounds__(256) void k_gemm_moe_out(const _Float16* __restrict__ A,
                                                      const _Float16* __restrict__ BT,
                                                      const float* __restrict__ bias,
                                                      const float* __restrict__ gates,
                                                      int e, float* __restrict__ out,
                                                      int N, int K) {
  __shared__ _Float16 smem[SMEM_HALVES];
  const int lane = threadIdx.x & 31, wave = threadIdx.x >> 5;
  const int m0 = blockIdx.y * 128;
  const int n0 = blockIdx.x * 256;
  v8f acc[4][4];
  gemm_core(A, BT, K, m0, n0, smem, acc);
  const int m0w = m0 + (wave & 1) * 64;
  const int n0w = n0 + (wave >> 1) * 64;
  const int rl = lane & 15, rh = (lane >> 4) << 3;
  float gv[4][8];
  #pragma unroll
  for (int i = 0; i < 4; i++)
    #pragma unroll
    for (int r = 0; r < 8; r++)
      gv[i][r] = gates[(size_t)(m0w + i * 16 + r + rh) * NE + e];
  #pragma unroll
  for (int j = 0; j < 4; j++) {
    const int n = n0w + j * 16 + rl;
    const float bv = bias[n];
    #pragma unroll
    for (int i = 0; i < 4; i++)
      #pragma unroll
      for (int r = 0; r < 8; r++) {
        const int m = m0w + i * 16 + r + rh;
        const float v = (acc[i][j][r] + bv) * gv[i][r];
        if (e == 0) out[(size_t)m * N + n] = v;
        else        out[(size_t)m * N + n] += v;
      }
  }
}

// ---------------- router: logits, softmax, top-2 gates ----------------------

__global__ __launch_bounds__(256) void k_router(const _Float16* __restrict__ y,
                                                const float* __restrict__ wr,
                                                float* __restrict__ gates,
                                                float* __restrict__ diag) {
  const int t = blockIdx.x * 256 + threadIdx.x;
  if (t >= TKN) return;
  float lg[NE] = {0.f, 0.f, 0.f, 0.f, 0.f, 0.f, 0.f, 0.f};
  const _Float16* yr = y + (size_t)t * DM;
  for (int d = 0; d < DM; d++) {
    const float xv = (float)yr[d];
    #pragma unroll
    for (int e = 0; e < NE; e++) lg[e] += xv * wr[d * NE + e];
  }
  float m = lg[0];
  #pragma unroll
  for (int e = 1; e < NE; e++) m = fmaxf(m, lg[e]);
  float p[NE], s = 0.f;
  #pragma unroll
  for (int e = 0; e < NE; e++) { p[e] = __expf(lg[e] - m); s += p[e]; }
  const float inv = 1.f / s;
  #pragma unroll
  for (int e = 0; e < NE; e++) p[e] *= inv;
  int i1 = 0;
  #pragma unroll
  for (int e = 1; e < NE; e++) if (p[e] > p[i1]) i1 = e;   // first max on ties
  int i2 = (i1 == 0) ? 1 : 0;
  #pragma unroll
  for (int e = 0; e < NE; e++) if (e != i1 && p[e] > p[i2]) i2 = e;
  const float den = p[i1] + p[i2];
  #pragma unroll
  for (int e = 0; e < NE; e++) {
    diag[(size_t)t * NE + e]  = p[e];
    gates[(size_t)t * NE + e] = (e == i1) ? p[i1] / den : (e == i2) ? p[i2] / den : 0.f;
  }
}

// ---------------- host-side orchestration -----------------------------------

extern "C" void kernel_launch(void* const* d_in, const int* in_sizes, int n_in,
                              void* d_out, int out_size, void* d_ws, size_t ws_size,
                              hipStream_t stream) {
  const float* x      = (const float*)d_in[0];
  const float* w_qkv  = (const float*)d_in[1];
  const float* b_qkv  = (const float*)d_in[2];
  const float* w_o    = (const float*)d_in[3];
  const float* b_o    = (const float*)d_in[4];
  const float* w_rtr  = (const float*)d_in[5];
  const float* w1     = (const float*)d_in[6];
  const float* b1     = (const float*)d_in[7];
  const float* w2     = (const float*)d_in[8];
  const float* b2     = (const float*)d_in[9];
  float* out  = (float*)d_out;               // [4096,768]
  float* diag = out + (size_t)TKN * DM;      // [4096,8]

  char* p = (char*)d_ws;
  auto take = [&](size_t bytes) { char* r = p; p += bytes; return r; };
  _Float16* x_h    = (_Float16*)take((size_t)TKN * DM * 2);
  _Float16* wqkvT  = (_Float16*)take((size_t)NQKV * DM * 2);
  _Float16* woT    = (_Float16*)take((size_t)DM * DM * 2);
  _Float16* w1T    = (_Float16*)take((size_t)NE * FF * DM * 2);
  _Float16* w2T    = (_Float16*)take((size_t)NE * DM * FF * 2);
  _Float16* q_h    = (_Float16*)take((size_t)NB * NH * SEQ * HD * 2);
  _Float16* k_h    = (_Float16*)take((size_t)NB * NH * SEQ * HD * 2);
  _Float16* vT_h   = (_Float16*)take((size_t)NB * NH * HD * SEQ * 2);
  _Float16* ctx_h  = (_Float16*)take((size_t)TKN * DM * 2);
  _Float16* y_h    = (_Float16*)take((size_t)TKN * DM * 2);
  _Float16* h_h    = (_Float16*)take((size_t)TKN * FF * 2);
  float*    gates  = (float*)take((size_t)TKN * NE * 4);
  (void)ws_size; (void)in_sizes; (void)n_in; (void)out_size;

  {
    int n = TKN * DM;
    k_f32_to_f16<<<(n + 255) / 256, 256, 0, stream>>>(x, x_h, n);
  }
  k_transpose_f16<<<(DM * NQKV + 255) / 256, 256, 0, stream>>>(w_qkv, wqkvT, DM, NQKV);
  k_transpose_f16<<<(DM * DM + 255) / 256, 256, 0, stream>>>(w_o, woT, DM, DM);
  for (int e = 0; e < NE; e++) {
    k_transpose_f16<<<(DM * FF + 255) / 256, 256, 0, stream>>>(
        w1 + (size_t)e * DM * FF, w1T + (size_t)e * FF * DM, DM, FF);
    k_transpose_f16<<<(FF * DM + 255) / 256, 256, 0, stream>>>(
        w2 + (size_t)e * FF * DM, w2T + (size_t)e * DM * FF, FF, DM);
  }

  // QKV projection (M=4096, N=2304, K=768) with head scatter
  k_gemm_qkv<<<dim3(NQKV / 256, TKN / 128), 256, 0, stream>>>(
      x_h, wqkvT, b_qkv, q_h, k_h, vT_h);

  // flash attention: 48 (b,h) x 8 q-blocks of 128 rows
  k_attention<<<dim3(NB * NH, SEQ / 128), 256, 0, stream>>>(q_h, k_h, vT_h, ctx_h);

  // output projection (M=4096, N=768, K=768)
  k_gemm_f16out<<<dim3(DM / 256, TKN / 128), 256, 0, stream>>>(
      ctx_h, woT, b_o, y_h, DM, DM, 0);

  // router probs + top-2 renormalized gates + diagnostics
  k_router<<<TKN / 256, 256, 0, stream>>>(y_h, w_rtr, gates, diag);

  // per-expert dense FFN (sequential on stream; h buffer reused)
  for (int e = 0; e < NE; e++) {
    k_gemm_f16out<<<dim3(FF / 256, TKN / 128), 256, 0, stream>>>(
        y_h, w1T + (size_t)e * FF * DM, b1 + (size_t)e * FF, h_h, FF, DM, 1);
    k_gemm_moe_out<<<dim3(DM / 256, TKN / 128), 256, 0, stream>>>(
        h_h, w2T + (size_t)e * DM * FF, b2 + (size_t)e * DM, gates, e, out, DM, FF);
  }
}
